// PointNetPP_67010079752486
// MI455X (gfx1250) — matrix operand order repaired and checked by reading, not compiled, not run on hardware
//
#include <hip/hip_runtime.h>

// ---------------------------------------------------------------------------
// Types for CDNA5 WMMA (wave32): v_wmma_f32_16x16x32_bf16
// ---------------------------------------------------------------------------
typedef __attribute__((ext_vector_type(16))) __bf16 v16bf;
typedef __attribute__((ext_vector_type(8)))  float  v8f;

union FragU { uint4 u[2]; v16bf v; };

__device__ __forceinline__ unsigned short f2bf_bits(float f) {
  union { float f; unsigned u; } v; v.f = f;
  // round-to-nearest-even f32 -> bf16
  return (unsigned short)((v.u + 0x7FFFu + ((v.u >> 16) & 1u)) >> 16);
}

// ---------------------------------------------------------------------------
// Pack A[M,K] (f32, row-major) into per-lane WMMA A-fragments (bf16).
// Fragment layout (CDNA5 ISA, 16-bit A 16x32):
//   lane = hi*16 + m(0..15); elem e (vgpr v=e/2, pair p=e&1):
//   K = chunk*32 + (v>=4 ? 16 : 0) + hi*8 + 2*(v&3) + p
// Output: PA[((mtile*nchunks + chunk)*32 + lane)*16 + e]  (16 ushorts = 32B/lane)
// Requires M % 16 == 0 (true for every GEMM in this net). K ragged -> zero pad.
// ---------------------------------------------------------------------------
__global__ void pack_a_kernel(const float* __restrict__ A, unsigned short* __restrict__ PA,
                              int K, int nchunks, int total) {
  const int t = blockIdx.x * 256 + threadIdx.x;
  if (t >= total) return;                       // total = mtiles*nchunks*32
  const int lane = t & 31;
  const int q    = (t >> 5) % nchunks;
  const int mt   = (t >> 5) / nchunks;
  const int hi   = lane >> 4;
  const int row  = mt * 16 + (lane & 15);
  const float* Ar = A + (size_t)row * K;
  union { uint4 u[2]; unsigned short e[16]; } frag;
#pragma unroll
  for (int e = 0; e < 16; ++e) {
    const int vv = e >> 1, p = e & 1;
    const int ka = q * 32 + ((vv & 4) ? 16 : 0) + hi * 8 + ((vv & 3) << 1) + p;
    const int kc = ka < K ? ka : K - 1;         // clamp address (always legal)
    float f = Ar[kc];
    f = (ka < K) ? f : 0.0f;                    // branchless zero-pad
    frag.e[e] = f2bf_bits(f);
  }
  uint4* o = (uint4*)(PA + (size_t)t * 16);
  o[0] = frag.u[0];
  o[1] = frag.u[1];
}

// ---------------------------------------------------------------------------
// Pack W[K,N] (f32, row-major) into per-lane WMMA B-fragments (bf16).
// Fragment layout (16-bit B 32x16): lane = hi*16 + n(0..15); elem e:
//   K = chunk*32 + (e>=8 ? 16 : 0) + hi*8 + (e&7)
// Output: PB[((ntile*nchunks + chunk)*32 + lane)*16 + e]
// ---------------------------------------------------------------------------
__global__ void pack_b_kernel(const float* __restrict__ W, unsigned short* __restrict__ PB,
                              int K, int N, int nchunks, int total) {
  const int t = blockIdx.x * 256 + threadIdx.x;
  if (t >= total) return;                       // total = ntiles*nchunks*32
  const int lane = t & 31;
  const int q    = (t >> 5) % nchunks;
  const int nt   = (t >> 5) / nchunks;
  const int hi   = lane >> 4;
  const int col  = nt * 16 + (lane & 15);
  const int colc = col < N ? col : N - 1;
  union { uint4 u[2]; unsigned short e[16]; } frag;
#pragma unroll
  for (int e = 0; e < 16; ++e) {
    const int kb = q * 32 + ((e & 8) ? 16 : 0) + hi * 8 + (e & 7);
    const int kc = kb < K ? kb : K - 1;
    float f = W[(size_t)kc * N + colc];
    f = (kb < K && col < N) ? f : 0.0f;
    frag.e[e] = f2bf_bits(f);
  }
  uint4* o = (uint4*)(PB + (size_t)t * 16);
  o[0] = frag.u[0];
  o[1] = frag.u[1];
}

// ---------------------------------------------------------------------------
// GEMM on pre-packed fragments: out[M,N] = A*W + bias.
// 128 threads = 4 waves (2x2). Each wave owns a 64(M) x 32(N) output block:
// 4 M-tiles x 2 N-tiles -> 8 WMMA per K-chunk from 6 b128-pair fragment loads
// (A frags reused across N-tiles, B frags reused across 4 M-tiles).
// Next chunk is prefetched (global_prefetch) under a wave-uniform guard.
// ---------------------------------------------------------------------------
__global__ __launch_bounds__(128) void gemm_wmma_packed(
    const unsigned short* __restrict__ PA, const unsigned short* __restrict__ PB,
    const float* __restrict__ bias, float* __restrict__ out,
    int N, int nchunks, int mtiles, int ntiles) {
  const int lane = threadIdx.x & 31;
  const int wv   = threadIdx.x >> 5;
  const int wm   = wv >> 1, wn = wv & 1;
  const int bmBase = blockIdx.x * 8 + wm * 4;   // first of 4 M-tiles
  const int bnBase = blockIdx.y * 4 + wn * 2;   // first of 2 N-tiles

  int tm[4], tn[2];
#pragma unroll
  for (int i = 0; i < 4; ++i) { const int v = bmBase + i; tm[i] = v < mtiles ? v : mtiles - 1; }
#pragma unroll
  for (int i = 0; i < 2; ++i) { const int v = bnBase + i; tn[i] = v < ntiles ? v : ntiles - 1; }

  const uint4* A4 = (const uint4*)PA;
  const uint4* B4 = (const uint4*)PB;

  v8f acc[4][2];
#pragma unroll
  for (int i = 0; i < 4; ++i)
#pragma unroll
    for (int j = 0; j < 2; ++j) acc[i][j] = (v8f){0.f,0.f,0.f,0.f,0.f,0.f,0.f,0.f};

  for (int q = 0; q < nchunks; ++q) {
    FragU a[4], b[2];
#pragma unroll
    for (int i = 0; i < 4; ++i) {
      const size_t ia = ((size_t)(tm[i] * nchunks + q) * 32 + lane) * 2;
      a[i].u[0] = A4[ia]; a[i].u[1] = A4[ia + 1];
    }
#pragma unroll
    for (int j = 0; j < 2; ++j) {
      const size_t ib = ((size_t)(tn[j] * nchunks + q) * 32 + lane) * 2;
      b[j].u[0] = B4[ib]; b[j].u[1] = B4[ib + 1];
    }
    if (q + 1 < nchunks) {  // wave-uniform: EXEC stays all-ones
      __builtin_prefetch(&A4[((size_t)(tm[0] * nchunks + q + 1) * 32 + lane) * 2], 0, 1);
      __builtin_prefetch(&B4[((size_t)(tn[0] * nchunks + q + 1) * 32 + lane) * 2], 0, 1);
    }
#pragma unroll
    for (int i = 0; i < 4; ++i) {
#pragma unroll
      for (int j = 0; j < 2; ++j) {
        acc[i][j] = __builtin_amdgcn_wmma_f32_16x16x32_bf16(
            false, a[i].v, false, b[j].v, (short)0, acc[i][j], false, false);
      }
    }
  }

  // Epilogue. D layout: lane = hi*16 + n(0..15); vgpr r -> M = hi*8 + r.
  const int hi = lane >> 4, l16 = lane & 15;
#pragma unroll
  for (int i = 0; i < 4; ++i) {
#pragma unroll
    for (int j = 0; j < 2; ++j) {
      const int bm = bmBase + i, bn = bnBase + j;
      if (bm < mtiles && bn < ntiles) {
        const int col = bn * 16 + l16;
        if (col < N) {
          const float bs = bias[col];
#pragma unroll
          for (int r = 0; r < 8; ++r) {
            const int row = bm * 16 + hi * 8 + r;
            out[(size_t)row * N + col] = acc[i][j][r] + bs;
          }
        }
      }
    }
  }
}

// ---------------------------------------------------------------------------
// Farthest point sampling. One block per batch. dist = scratch [B*N].
// ---------------------------------------------------------------------------
__global__ void fps_kernel(const float* __restrict__ xyz, int* __restrict__ idx_out,
                           float* __restrict__ dist, int N, int S) {
  const int b = blockIdx.x, tid = threadIdx.x;
  const float* X = xyz + (size_t)b * N * 3;
  float* D = dist + (size_t)b * N;
  __shared__ float smax[256];
  __shared__ int   simax[256];
  for (int i = tid; i < N; i += 256) D[i] = 1e10f;
  __syncthreads();
  int far = 0;
  for (int s = 0; s < S; ++s) {
    if (tid == 0) idx_out[b * S + s] = far;
    const float cx = X[far * 3 + 0], cy = X[far * 3 + 1], cz = X[far * 3 + 2];
    float best = -1.0f; int bi = 0;
    for (int i = tid; i < N; i += 256) {
      const float dx = X[i * 3 + 0] - cx, dy = X[i * 3 + 1] - cy, dz = X[i * 3 + 2] - cz;
      const float d = dx * dx + dy * dy + dz * dz;
      const float dm = fminf(D[i], d);
      D[i] = dm;
      if (dm > best) { best = dm; bi = i; }
    }
    smax[tid] = best; simax[tid] = bi;
    __syncthreads();
    for (int st = 128; st > 0; st >>= 1) {
      if (tid < st) {
        if (smax[tid + st] > smax[tid] ||
            (smax[tid + st] == smax[tid] && simax[tid + st] < simax[tid])) {
          smax[tid] = smax[tid + st]; simax[tid] = simax[tid + st];
        }
      }
      __syncthreads();
    }
    far = simax[0];
    __syncthreads();
  }
}

// dst[b,s,:] = src[b, idx[b,s], :]   (3 channels)
__global__ void gather3_kernel(const float* __restrict__ src, const int* __restrict__ idx,
                               float* __restrict__ dst, int N, int S, int total) {
  const int t = blockIdx.x * 256 + threadIdx.x;
  if (t >= total) return;                 // total = B*S
  const int b = t / S;
  int j = idx[t];
  if (j < 0) j = 0; if (j >= N) j = N - 1;
  const float* p = src + ((size_t)b * N + j) * 3;
  float* o = dst + (size_t)t * 3;
  o[0] = p[0]; o[1] = p[1]; o[2] = p[2];
}

// Ball query: first K indices (ascending) with dist^2 <= r2, padded with first.
__global__ void query_ball_kernel(const float* __restrict__ xyz,
                                  const float* __restrict__ new_xyz,
                                  int* __restrict__ idx, int N, int S, int K,
                                  float r2, int total) {
  const int t = blockIdx.x * 256 + threadIdx.x;
  if (t >= total) return;                 // total = B*S
  const int b = t / S;
  const float qx = new_xyz[(size_t)t * 3 + 0];
  const float qy = new_xyz[(size_t)t * 3 + 1];
  const float qz = new_xyz[(size_t)t * 3 + 2];
  const float* X = xyz + (size_t)b * N * 3;
  int* o = idx + (size_t)t * K;
  int cnt = 0;
  for (int i = 0; i < N && cnt < K; ++i) {
    const float dx = X[i * 3 + 0] - qx, dy = X[i * 3 + 1] - qy, dz = X[i * 3 + 2] - qz;
    if (dx * dx + dy * dy + dz * dz <= r2) o[cnt++] = i;
  }
  const int first = (cnt > 0) ? o[0] : 0;
  for (int k = cnt; k < K; ++k) o[k] = first;
}

// Build grouped rows: [points[b,idx,:Cp] , xyz[b,idx,:]-new_xyz[b,s,:]]
__global__ void group_kernel(const float* __restrict__ xyz, const float* __restrict__ points,
                             const float* __restrict__ new_xyz, const int* __restrict__ idx,
                             float* __restrict__ out, int N, int S, int K, int Cp, int total) {
  const int t = blockIdx.x * 256 + threadIdx.x;
  if (t >= total) return;                 // total = B*S*K
  const int bs = t / K;
  const int b  = bs / S;
  int j = idx[t];
  if (j < 0) j = 0; if (j >= N) j = N - 1;
  const int C = Cp + 3;
  float* o = out + (size_t)t * C;
  if (points != nullptr) {
    const float* p = points + ((size_t)b * N + j) * Cp;
    for (int c = 0; c < Cp; ++c) o[c] = p[c];
  }
  const float* xp = xyz + ((size_t)b * N + j) * 3;
  const float* nq = new_xyz + (size_t)bs * 3;
  o[Cp + 0] = xp[0] - nq[0];
  o[Cp + 1] = xp[1] - nq[1];
  o[Cp + 2] = xp[2] - nq[2];
}

// In-place batchnorm(+relu) over all rows, per channel. Block per channel.
__global__ void bn_relu_kernel(float* __restrict__ x, const float* __restrict__ g,
                               const float* __restrict__ beta, int M, int C) {
  const int c = blockIdx.x, tid = threadIdx.x;
  __shared__ float ssum[256], ssq[256];
  float s = 0.f, q = 0.f;
  for (int i = tid; i < M; i += 256) {
    const float v = x[(size_t)i * C + c];
    s += v; q += v * v;
  }
  ssum[tid] = s; ssq[tid] = q;
  __syncthreads();
  for (int st = 128; st > 0; st >>= 1) {
    if (tid < st) { ssum[tid] += ssum[tid + st]; ssq[tid] += ssq[tid + st]; }
    __syncthreads();
  }
  const float mean = ssum[0] / (float)M;
  const float var  = ssq[0] / (float)M - mean * mean;
  const float inv  = rsqrtf(var + 1e-5f);
  const float gg = g[c], bb = beta[c];
  for (int i = tid; i < M; i += 256) {
    const size_t o = (size_t)i * C + c;
    const float v = (x[o] - mean) * inv * gg + bb;
    x[o] = v > 0.f ? v : 0.f;
  }
}

// out[r, colOff+c] = max_k in[(r*K+k), c]
__global__ void maxpool_kernel(const float* __restrict__ in, float* __restrict__ out,
                               int K, int C, int dstStride, int colOff, int total) {
  const int t = blockIdx.x * 256 + threadIdx.x;
  if (t >= total) return;                 // total = R*C
  const int r = t / C, c = t % C;
  const float* p = in + ((size_t)r * K) * C + c;
  float m = p[0];
  for (int k = 1; k < K; ++k) { const float v = p[(size_t)k * C]; m = v > m ? v : m; }
  out[(size_t)r * dstStride + colOff + c] = m;
}

// dst[r, colOff+c] = src[r, c]
__global__ void copy_cols_kernel(const float* __restrict__ src, float* __restrict__ dst,
                                 int Csrc, int dstStride, int colOff, int total) {
  const int t = blockIdx.x * 256 + threadIdx.x;
  if (t >= total) return;                 // total = R*Csrc
  const int r = t / Csrc, c = t % Csrc;
  dst[(size_t)r * dstStride + colOff + c] = src[t];
}

// dst[(b*Rper+r), colOff+c] = src[b, c]
__global__ void broadcast_kernel(const float* __restrict__ src, float* __restrict__ dst,
                                 int Rper, int C, int dstStride, int colOff, int total) {
  const int t = blockIdx.x * 256 + threadIdx.x;
  if (t >= total) return;                 // total = B*Rper*C
  const int c = t % C;
  const int br = t / C;
  const int b = br / Rper;
  dst[(size_t)br * dstStride + colOff + c] = src[(size_t)b * C + c];
}

// 3-NN inverse-distance interpolation. One thread per (b, n).
__global__ void interp_kernel(const float* __restrict__ xyz1, const float* __restrict__ xyz2,
                              const float* __restrict__ pts2, float* __restrict__ dst,
                              int N, int S, int C, int dstStride, int colOff, int total) {
  const int t = blockIdx.x * 256 + threadIdx.x;
  if (t >= total) return;                 // total = B*N
  const int b = t / N;
  const float qx = xyz1[(size_t)t * 3 + 0];
  const float qy = xyz1[(size_t)t * 3 + 1];
  const float qz = xyz1[(size_t)t * 3 + 2];
  const float* X2 = xyz2 + (size_t)b * S * 3;
  float d0 = 3.4e38f, d1 = 3.4e38f, d2 = 3.4e38f;
  int i0 = 0, i1 = 0, i2 = 0;
  for (int i = 0; i < S; ++i) {
    const float dx = X2[i * 3 + 0] - qx, dy = X2[i * 3 + 1] - qy, dz = X2[i * 3 + 2] - qz;
    const float d = dx * dx + dy * dy + dz * dz;
    if (d < d0)      { d2 = d1; i2 = i1; d1 = d0; i1 = i0; d0 = d; i0 = i; }
    else if (d < d1) { d2 = d1; i2 = i1; d1 = d;  i1 = i; }
    else if (d < d2) { d2 = d;  i2 = i; }
  }
  float w0 = 1.f / (d0 + 1e-8f), w1 = 1.f / (d1 + 1e-8f), w2 = 1.f / (d2 + 1e-8f);
  const float ws = w0 + w1 + w2;
  w0 /= ws; w1 /= ws; w2 /= ws;
  const float* P = pts2 + (size_t)b * S * C;
  float* o = dst + (size_t)t * dstStride + colOff;
  for (int c = 0; c < C; ++c)
    o[c] = w0 * P[(size_t)i0 * C + c] + w1 * P[(size_t)i1 * C + c] + w2 * P[(size_t)i2 * C + c];
}

// dst[b, c, n] = src[(b*N+n), c]
__global__ void transpose_kernel(const float* __restrict__ src, float* __restrict__ dst,
                                 int Nn, int C, int total) {
  const int t = blockIdx.x * 256 + threadIdx.x;
  if (t >= total) return;                 // total = B*N*C
  const int c = t % C;
  const int bn = t / C;
  const int n = bn % Nn;
  const int b = bn / Nn;
  dst[((size_t)b * C + c) * Nn + n] = src[t];
}

// ---------------------------------------------------------------------------
// Host orchestration
// ---------------------------------------------------------------------------
static inline int cdiv(long a, int b) { return (int)((a + b - 1) / b); }

struct Layer { const float *W, *b, *g, *beta; int cin, cout; };

extern "C" void kernel_launch(void* const* d_in, const int* in_sizes, int n_in,
                              void* d_out, int out_size, void* d_ws, size_t ws_size,
                              hipStream_t stream) {
  (void)in_sizes; (void)n_in; (void)out_size; (void)ws_size;
  const int B = 8, N0 = 4096;

  const float* xyz = (const float*)d_in[0];
  int cur = 1;
  auto nextp = [&]() { return (const float*)d_in[cur++]; };
  auto getLayer = [&](int cin, int cout) {
    Layer L; L.W = nextp(); L.b = nextp(); L.g = nextp(); L.beta = nextp();
    L.cin = cin; L.cout = cout; return L;
  };

  // params in setup_inputs() insertion order: sa1, sa2, sa3, fp3, fp2, fp1, conv1, embed
  Layer sa1[3][3];
  const int ch1[3][3] = {{32, 32, 64}, {64, 64, 128}, {64, 96, 128}};
  for (int j = 0; j < 3; ++j) { int ci = 3; for (int l = 0; l < 3; ++l) { sa1[j][l] = getLayer(ci, ch1[j][l]); ci = ch1[j][l]; } }
  Layer sa2l[2][3];
  const int ch2[2][3] = {{128, 128, 256}, {128, 196, 256}};
  for (int j = 0; j < 2; ++j) { int ci = 323; for (int l = 0; l < 3; ++l) { sa2l[j][l] = getLayer(ci, ch2[j][l]); ci = ch2[j][l]; } }
  Layer sa3l[3];
  { int ci = 515; const int ch[3] = {256, 512, 1024}; for (int l = 0; l < 3; ++l) { sa3l[l] = getLayer(ci, ch[l]); ci = ch[l]; } }
  Layer fp3l[2]; { int ci = 1536; const int ch[2] = {256, 256}; for (int l = 0; l < 2; ++l) { fp3l[l] = getLayer(ci, ch[l]); ci = ch[l]; } }
  Layer fp2l[2]; { int ci = 576;  const int ch[2] = {256, 128}; for (int l = 0; l < 2; ++l) { fp2l[l] = getLayer(ci, ch[l]); ci = ch[l]; } }
  Layer fp1l[2]; { int ci = 131;  const int ch[2] = {128, 128}; for (int l = 0; l < 2; ++l) { fp1l[l] = getLayer(ci, ch[l]); ci = ch[l]; } }
  Layer conv1 = getLayer(128, 128);
  const float* embW = nextp();
  const float* embB = nextp();

  // workspace bump allocator
  char* wsb = (char*)d_ws;
  size_t off = 0;
  auto allocf = [&](size_t elems) { float* p = (float*)(wsb + off); off += ((elems * 4 + 255) / 256) * 256; return p; };
  auto alloci = [&](size_t elems) { int*   p = (int*)  (wsb + off); off += ((elems * 4 + 255) / 256) * 256; return p; };
  auto allocs = [&](size_t elems) { unsigned short* p = (unsigned short*)(wsb + off); off += ((elems * 2 + 255) / 256) * 256; return p; };

  float* dist      = allocf((size_t)B * N0);
  int*   idx1      = alloci((size_t)B * 512);
  float* l1_xyz    = allocf((size_t)B * 512 * 3);
  int*   idx2      = alloci((size_t)B * 128);
  float* l2_xyz    = allocf((size_t)B * 128 * 3);
  int*   idxK      = alloci((size_t)B * 512 * 128);
  float* l1_points = allocf((size_t)B * 512 * 320);
  float* l2_points = allocf((size_t)B * 128 * 512);
  float* l3_points = allocf((size_t)B * 1024);
  float* l2f       = allocf((size_t)B * 128 * 256);
  float* l1f       = allocf((size_t)B * 512 * 128);
  const size_t PP  = (size_t)B * 512 * 128 * 128;   // largest f32 intermediate
  float* bufA      = allocf(PP);
  float* bufB      = allocf(PP);
  // packed fragment buffers: PA needs mtiles*nchunks*512 bf16 (max = M*Kpad)
  unsigned short* packA = allocs((size_t)32768 * 4 * 512);  // 524288x128 worst case
  unsigned short* packB = allocs((size_t)64 * 48 * 512);    // 1536x1024 worst case

  // one GEMM layer: pack A + pack B + wmma gemm (+bias)
  auto run_gemm = [&](const float* x, const float* W, const float* bias, float* y,
                      int M, int K, int N) {
    const int mtiles = (M + 15) / 16;
    const int ntiles = (N + 15) / 16;
    const int nchunks = (K + 31) / 32;
    const long ta = (long)mtiles * nchunks * 32;
    const long tb = (long)ntiles * nchunks * 32;
    pack_a_kernel<<<cdiv(ta, 256), 256, 0, stream>>>(x, packA, K, nchunks, (int)ta);
    pack_b_kernel<<<cdiv(tb, 256), 256, 0, stream>>>(W, packB, K, N, nchunks, (int)tb);
    dim3 g(cdiv(mtiles, 8), cdiv(ntiles, 4));
    gemm_wmma_packed<<<g, 128, 0, stream>>>(packA, packB, bias, y, N, nchunks, mtiles, ntiles);
  };

  auto run_mlp = [&](float* x, float* y, int M, const Layer* Ls, int nl) -> float* {
    for (int l = 0; l < nl; ++l) {
      const Layer& L = Ls[l];
      run_gemm(x, L.W, L.b, y, M, L.cin, L.cout);
      bn_relu_kernel<<<L.cout, 256, 0, stream>>>(y, L.g, L.beta, M, L.cout);
      float* t = x; x = y; y = t;
    }
    return x;  // final result
  };

  // ---------------- SA1 (l0 -> l1): npoint=512, cin=3 ----------------
  fps_kernel<<<B, 256, 0, stream>>>(xyz, idx1, dist, N0, 512);
  gather3_kernel<<<cdiv((long)B * 512, 256), 256, 0, stream>>>(xyz, idx1, l1_xyz, N0, 512, B * 512);
  {
    const int   K1[3] = {32, 64, 128};
    const float R1[3] = {0.1f, 0.2f, 0.4f};
    int colOff = 0;
    for (int j = 0; j < 3; ++j) {
      query_ball_kernel<<<cdiv((long)B * 512, 256), 256, 0, stream>>>(
          xyz, l1_xyz, idxK, N0, 512, K1[j], R1[j] * R1[j], B * 512);
      const int M = B * 512 * K1[j];
      group_kernel<<<cdiv(M, 256), 256, 0, stream>>>(xyz, nullptr, l1_xyz, idxK, bufA, N0, 512, K1[j], 0, M);
      float* res = run_mlp(bufA, bufB, M, sa1[j], 3);
      const int cl = ch1[j][2];
      maxpool_kernel<<<cdiv((long)B * 512 * cl, 256), 256, 0, stream>>>(
          res, l1_points, K1[j], cl, 320, colOff, B * 512 * cl);
      colOff += cl;
    }
  }

  // ---------------- SA2 (l1 -> l2): npoint=128, cin=323 ----------------
  fps_kernel<<<B, 256, 0, stream>>>(l1_xyz, idx2, dist, 512, 128);
  gather3_kernel<<<cdiv((long)B * 128, 256), 256, 0, stream>>>(l1_xyz, idx2, l2_xyz, 512, 128, B * 128);
  {
    const int   K2[2] = {64, 128};
    const float R2[2] = {0.4f, 0.8f};
    int colOff = 0;
    for (int j = 0; j < 2; ++j) {
      query_ball_kernel<<<cdiv((long)B * 128, 256), 256, 0, stream>>>(
          l1_xyz, l2_xyz, idxK, 512, 128, K2[j], R2[j] * R2[j], B * 128);
      const int M = B * 128 * K2[j];
      group_kernel<<<cdiv(M, 256), 256, 0, stream>>>(l1_xyz, l1_points, l2_xyz, idxK, bufA, 512, 128, K2[j], 320, M);
      float* res = run_mlp(bufA, bufB, M, sa2l[j], 3);
      maxpool_kernel<<<cdiv((long)B * 128 * 256, 256), 256, 0, stream>>>(
          res, l2_points, K2[j], 256, 512, colOff, B * 128 * 256);
      colOff += 256;
    }
  }

  // ---------------- SA3 (global): cin=515, pool over the 128 points ----------------
  {
    const int M = B * 128;
    copy_cols_kernel<<<cdiv((long)M * 3, 256), 256, 0, stream>>>(l2_xyz, bufA, 3, 515, 0, M * 3);
    copy_cols_kernel<<<cdiv((long)M * 512, 256), 256, 0, stream>>>(l2_points, bufA, 512, 515, 3, M * 512);
    float* res = run_mlp(bufA, bufB, M, sa3l, 3);
    maxpool_kernel<<<cdiv((long)B * 1024, 256), 256, 0, stream>>>(res, l3_points, 128, 1024, 1024, 0, B * 1024);
  }

  // ---------------- FP3: broadcast l3 (S==1), concat with l2_points ----------------
  {
    const int M = B * 128;
    copy_cols_kernel<<<cdiv((long)M * 512, 256), 256, 0, stream>>>(l2_points, bufA, 512, 1536, 0, M * 512);
    broadcast_kernel<<<cdiv((long)M * 1024, 256), 256, 0, stream>>>(l3_points, bufA, 128, 1024, 1536, 512, M * 1024);
    float* res = run_mlp(bufA, bufB, M, fp3l, 2);
    copy_cols_kernel<<<cdiv((long)M * 256, 256), 256, 0, stream>>>(res, l2f, 256, 256, 0, M * 256);
  }

  // ---------------- FP2: interp l2 -> l1, concat l1_points ----------------
  {
    const int M = B * 512;
    copy_cols_kernel<<<cdiv((long)M * 320, 256), 256, 0, stream>>>(l1_points, bufA, 320, 576, 0, M * 320);
    interp_kernel<<<cdiv((long)M, 256), 256, 0, stream>>>(l1_xyz, l2_xyz, l2f, bufA, 512, 128, 256, 576, 320, M);
    float* res = run_mlp(bufA, bufB, M, fp2l, 2);
    copy_cols_kernel<<<cdiv((long)M * 128, 256), 256, 0, stream>>>(res, l1f, 128, 128, 0, M * 128);
  }

  // ---------------- FP1: interp l1 -> l0, concat l0 xyz ----------------
  const int M0 = B * N0;
  {
    copy_cols_kernel<<<cdiv((long)M0 * 3, 256), 256, 0, stream>>>(xyz, bufA, 3, 131, 0, M0 * 3);
    interp_kernel<<<cdiv((long)M0, 256), 256, 0, stream>>>(xyz, l1_xyz, l1f, bufA, N0, 512, 128, 131, 3, M0);
    run_mlp(bufA, bufB, M0, fp1l, 2);  // 2 layers -> result lands back in bufA
  }

  // ---------------- conv1 (gemm + BN + relu), embed (gemm), transpose ----------------
  {
    run_gemm(bufA, conv1.W, conv1.b, bufB, M0, 128, 128);
    bn_relu_kernel<<<128, 256, 0, stream>>>(bufB, conv1.g, conv1.beta, M0, 128);
    run_gemm(bufB, embW, embB, bufA, M0, 128, 128);
    transpose_kernel<<<cdiv((long)M0 * 128, 256), 256, 0, stream>>>(bufA, (float*)d_out, N0, 128, M0 * 128);
  }
}
